// SelfAttention_61572651155726
// MI455X (gfx1250) — matrix operand order repaired and checked by reading, not compiled, not run on hardware
//
#include <hip/hip_runtime.h>
#include <hip/hip_bf16.h>

typedef _Float16 half_t;
typedef __attribute__((ext_vector_type(16))) _Float16 v16h;
typedef __attribute__((ext_vector_type(8)))  _Float16 v8h;
typedef __attribute__((ext_vector_type(8)))  float    v8f;

#define DIMD  1024
#define BATCH 4
#define SEQ   2048
#define BSEQ  (BATCH * SEQ)   // 8192

// ---------------------------------------------------------------------------
// gfx1250 async copy: global -> LDS, 16 bytes per lane, tracked by ASYNCcnt.
// ---------------------------------------------------------------------------
__device__ __forceinline__ void async_copy_b128(unsigned lds_off, const void* gptr)
{
  asm volatile("global_load_async_to_lds_b128 %0, %1, off"
               :: "v"(lds_off), "v"(gptr)
               : "memory");
}

template <int N>
__device__ __forceinline__ void wait_async_le()
{
#if __has_builtin(__builtin_amdgcn_s_wait_asynccnt)
  __builtin_amdgcn_s_wait_asynccnt(N);
#else
  asm volatile("s_wait_asynccnt %0" :: "i"(N) : "memory");
#endif
}

// ---------------------------------------------------------------------------
// Elementwise convert fp32 -> f16 (x4 vectorized). n must be a multiple of 4.
// ---------------------------------------------------------------------------
__global__ __launch_bounds__(256)
void convert_f32_f16(const float* __restrict__ in, half_t* __restrict__ out, long n)
{
  long i = ((long)blockIdx.x * 256 + threadIdx.x) * 4;
  if (i + 3 < n) {
    const float4 v = *(const float4*)(in + i);
    out[i + 0] = (half_t)v.x;
    out[i + 1] = (half_t)v.y;
    out[i + 2] = (half_t)v.z;
    out[i + 3] = (half_t)v.w;
  }
}

// ---------------------------------------------------------------------------
// Transpose + convert: W[K,N] fp32 (row-major) -> WT[N,K] f16 (row-major)
// ---------------------------------------------------------------------------
__global__ __launch_bounds__(256)
void transpose_f32_f16(const float* __restrict__ W, half_t* __restrict__ WT,
                       int K, int N)
{
  __shared__ float tile[32][33];
  const int n0 = blockIdx.x * 32;
  const int k0 = blockIdx.y * 32;
  const int tx = threadIdx.x & 31;
  const int ty = threadIdx.x >> 5;
  #pragma unroll
  for (int r = ty; r < 32; r += 8)
    tile[r][tx] = W[(long)(k0 + r) * N + (n0 + tx)];
  __syncthreads();
  #pragma unroll
  for (int r = ty; r < 32; r += 8)
    WT[(long)(n0 + r) * K + (k0 + tx)] = (half_t)tile[tx][r];
}

// ---------------------------------------------------------------------------
// WMMA GEMM:  C[M,N] = A[M,K] * BT[N,K]^T   (A,BT f16 row-major, fp32 accum)
//
// 256 threads = 8 wave32s; block tile 128x128; waves 4(M)x2(N); wave tile
// 32x64 = 2x4 WMMA 16x16x32 tiles. K staged in 32-wide slabs through a
// 3-deep LDS pipeline fed by global_load_async_to_lds_b128: two slabs in
// flight, steady-state wait is ASYNCcnt <= 4 (async loads complete in
// order), so copy latency is hidden behind two compute phases.
//
// MODE 0: qkv epilogue (+b_qkv; split Q(x1/32)/K/V^T)
// MODE 1: fp32 scores    MODE 2: f16 O    MODE 3: fp32 out + b_proj
// ---------------------------------------------------------------------------
template <int MODE>
__global__ __launch_bounds__(256)
void gemm_wmma(const half_t* __restrict__ A, const half_t* __restrict__ BT,
               float* __restrict__ Cf, half_t* __restrict__ Ch,
               const float* __restrict__ bias,
               int N, int K,
               long strideA, long strideB, long strideC,
               half_t* __restrict__ Qh, half_t* __restrict__ Kh,
               half_t* __restrict__ VhT)
{
  const int tid  = threadIdx.x;
  const int wave = tid >> 5;      // 0..7
  const int lane = tid & 31;
  const int wm   = wave >> 1;     // 0..3  (M direction)
  const int wn   = wave & 1;      // 0..1  (N direction)
  const int bm   = blockIdx.y * 128;
  const int bn   = blockIdx.x * 128;
  const int bz   = blockIdx.z;

  A  += (long)bz * strideA;
  BT += (long)bz * strideB;

  // Padded row stride 48 halves = 96 B keeps every b128 access 16B-aligned.
  // Triple-buffered: 3 x (12 KB + 12 KB) = 72 KB of the 320 KB WGP LDS.
  __shared__ __attribute__((aligned(32))) half_t As[3][128][48];
  __shared__ __attribute__((aligned(32))) half_t Bs[3][128][48];
  constexpr unsigned BUFB = 128u * 48u * sizeof(half_t);   // 12288 B

  v8f acc[2][4] = {};

  const int lrow = lane & 15;
  const int lhi  = lane >> 4;

  // ---- hoisted per-thread staging addresses (4 x b128 per slab) ----
  const int r0 = tid >> 2;        // 0..63
  const int c0 = tid & 3;         // 16-byte chunk within the 32-half slab row
  const half_t* gA0 = A  + (long)(bm + r0) * K + c0 * 8;
  const half_t* gA1 = gA0 + (long)64 * K;
  const half_t* gB0 = BT + (long)(bn + r0) * K + c0 * 8;
  const half_t* gB1 = gB0 + (long)64 * K;
  const unsigned lA0 = (unsigned)(uintptr_t)&As[0][r0][c0 * 8];
  const unsigned lA1 = (unsigned)(uintptr_t)&As[0][r0 + 64][c0 * 8];
  const unsigned lB0 = (unsigned)(uintptr_t)&Bs[0][r0][c0 * 8];
  const unsigned lB1 = (unsigned)(uintptr_t)&Bs[0][r0 + 64][c0 * 8];

  auto stage = [&](int buf, int k0) {
    const unsigned o = (unsigned)buf * BUFB;
    async_copy_b128(lA0 + o, gA0 + k0);
    async_copy_b128(lA1 + o, gA1 + k0);
    async_copy_b128(lB0 + o, gB0 + k0);
    async_copy_b128(lB1 + o, gB1 + k0);
  };

  auto compute = [&](int cur) {
    const half_t (*__restrict__ Ab)[48] = As[cur];
    const half_t (*__restrict__ Bb)[48] = Bs[cur];

    // A fragment (16x32): lane L(0-15) row M=L holds K 0..7 & 16..23,
    // lane L+16 holds K 8..15 & 24..31 (ISA 7.12.2).
    v16h afr[2];
    #pragma unroll
    for (int tm = 0; tm < 2; ++tm) {
      const int r = wm * 32 + tm * 16 + lrow;
      const v8h lo = *(const v8h*)&Ab[r][lhi * 8];
      const v8h hi = *(const v8h*)&Ab[r][16 + lhi * 8];
      afr[tm] = __builtin_shufflevector(lo, hi, 0, 1, 2, 3, 4, 5, 6, 7,
                                                8, 9, 10, 11, 12, 13, 14, 15);
    }
    // B fragment (32x16): lane L holds col N=L%16, K=(L/16)*8 + 0..7 & 16+(L/16)*8 + 0..7.
    v16h bfr[4];
    #pragma unroll
    for (int tn = 0; tn < 4; ++tn) {
      const int r = wn * 64 + tn * 16 + lrow;
      const v8h lo = *(const v8h*)&Bb[r][lhi * 8];
      const v8h hi = *(const v8h*)&Bb[r][16 + lhi * 8];
      bfr[tn] = __builtin_shufflevector(lo, hi, 0, 1, 2, 3, 4, 5, 6, 7,
                                                8, 9, 10, 11, 12, 13, 14, 15);
    }

    #pragma unroll
    for (int tm = 0; tm < 2; ++tm)
      #pragma unroll
      for (int tn = 0; tn < 4; ++tn)
        acc[tm][tn] = __builtin_amdgcn_wmma_f32_16x16x32_f16(
            false, afr[tm], false, bfr[tn], (short)0, acc[tm][tn], false, false);
  };

  const int nk = K >> 5;          // >= 32 at every call site
  stage(0, 0);
  stage(1, 32);

  int cur = 0;
  for (int kk = 0; kk < nk - 1; ++kk) {
    wait_async_le<4>();   // slab kk landed (in-order); slab kk+1 still in flight
    __syncthreads();      // all waves' copies landed; slab kk-1 reads consumed
    if (kk + 2 < nk) {
      const int nxt = (cur + 2 >= 3) ? cur - 1 : cur + 2;
      stage(nxt, (kk + 2) << 5);
    }
    compute(cur);
    cur = (cur == 2) ? 0 : cur + 1;
  }
  // last slab: drain everything
  wait_async_le<0>();
  __syncthreads();
  compute(cur);

  // Epilogue. C/D layout: VGPR j, lane L -> M = j + (L>>4)*8, N = L&15.
  #pragma unroll
  for (int tm = 0; tm < 2; ++tm) {
    #pragma unroll
    for (int tn = 0; tn < 4; ++tn) {
      #pragma unroll
      for (int j = 0; j < 8; ++j) {
        const int m = bm + wm * 32 + tm * 16 + lhi * 8 + j;
        const int n = bn + wn * 64 + tn * 16 + lrow;
        float v = acc[tm][tn][j];
        if constexpr (MODE == 0) {
          v += bias[n];
          const int seg = n >> 10;        // 0=q 1=k 2=v
          const int nl  = n & (DIMD - 1);
          const int b   = m >> 11;        // SEQ = 2048 rows per batch
          const int s   = m & (SEQ - 1);
          const long qi = ((long)b * SEQ + s) * DIMD + nl;
          if (seg == 0)      Qh[qi] = (half_t)(v * 0.03125f);   // fold 1/sqrt(D)
          else if (seg == 1) Kh[qi] = (half_t)v;
          else               VhT[((long)b * DIMD + nl) * SEQ + s] = (half_t)v;
        } else if constexpr (MODE == 1) {
          Cf[(long)bz * strideC + (long)m * N + n] = v;
        } else if constexpr (MODE == 2) {
          Ch[(long)bz * strideC + (long)m * N + n] = (half_t)v;
        } else {
          Cf[(long)m * N + n] = v + bias[n];
        }
      }
    }
  }
}

// ---------------------------------------------------------------------------
// Row softmax: scores fp32 [rows, 2048] -> P f16. One 256-thread block / row.
// ---------------------------------------------------------------------------
__global__ __launch_bounds__(256)
void softmax_rows(const float* __restrict__ S, half_t* __restrict__ P)
{
  const long row = blockIdx.x;
  const float* s = S + row * (long)SEQ;
  half_t* p = P + row * (long)SEQ;
  const int tid = threadIdx.x;

  float vals[8];
  float mx = -3.402823466e+38f;
  #pragma unroll
  for (int i = 0; i < 8; ++i) {
    vals[i] = s[tid + i * 256];
    mx = fmaxf(mx, vals[i]);
  }
  #pragma unroll
  for (int off = 16; off > 0; off >>= 1)
    mx = fmaxf(mx, __shfl_xor(mx, off, 32));

  __shared__ float redmx[8];
  __shared__ float redsum[8];
  if ((tid & 31) == 0) redmx[tid >> 5] = mx;
  __syncthreads();
  mx = redmx[0];
  #pragma unroll
  for (int i = 1; i < 8; ++i) mx = fmaxf(mx, redmx[i]);

  float sum = 0.f;
  #pragma unroll
  for (int i = 0; i < 8; ++i) {
    vals[i] = __expf(vals[i] - mx);
    sum += vals[i];
  }
  #pragma unroll
  for (int off = 16; off > 0; off >>= 1)
    sum += __shfl_xor(sum, off, 32);
  if ((tid & 31) == 0) redsum[tid >> 5] = sum;
  __syncthreads();
  sum = 0.f;
  #pragma unroll
  for (int i = 0; i < 8; ++i) sum += redsum[i];

  const float inv = 1.0f / sum;
  #pragma unroll
  for (int i = 0; i < 8; ++i)
    p[tid + i * 256] = (half_t)(vals[i] * inv);
}

// ---------------------------------------------------------------------------
// Host launch
// ---------------------------------------------------------------------------
extern "C" void kernel_launch(void* const* d_in, const int* in_sizes, int n_in,
                              void* d_out, int out_size, void* d_ws, size_t ws_size,
                              hipStream_t stream)
{
  const float* x      = (const float*)d_in[0];   // [4,2048,1024]
  const float* w_qkv  = (const float*)d_in[1];   // [1024,3072]
  const float* b_qkv  = (const float*)d_in[2];   // [3072]
  const float* w_proj = (const float*)d_in[3];   // [1024,1024]
  const float* b_proj = (const float*)d_in[4];   // [1024]
  float* out = (float*)d_out;                    // [4,2048,1024] fp32

  // Workspace carve (all sizes 256B-aligned by construction).
  char* ws = (char*)d_ws;
  half_t* xh     = (half_t*)ws; ws += (long)BSEQ * DIMD * 2;        // 16 MB
  half_t* wqkvT  = (half_t*)ws; ws += (long)3 * DIMD * DIMD * 2;    //  6 MB
  half_t* wprojT = (half_t*)ws; ws += (long)DIMD * DIMD * 2;        //  2 MB
  half_t* Qh     = (half_t*)ws; ws += (long)BSEQ * DIMD * 2;        // 16 MB
  half_t* Kh     = (half_t*)ws; ws += (long)BSEQ * DIMD * 2;        // 16 MB
  half_t* VhT    = (half_t*)ws; ws += (long)BSEQ * DIMD * 2;        // 16 MB
  float*  Sc     = (float*)ws;  ws += (long)BATCH * SEQ * SEQ * 4;  // 64 MB
  half_t* P      = (half_t*)ws; ws += (long)BATCH * SEQ * SEQ * 2;  // 32 MB
  half_t* Oh     = (half_t*)ws;                                     // 16 MB

  // 1) Precision conversion / weight transposition (one-shot, bandwidth-cheap).
  convert_f32_f16<<<dim3((BSEQ * DIMD) / 1024), 256, 0, stream>>>(
      x, xh, (long)BSEQ * DIMD);
  transpose_f32_f16<<<dim3(3 * DIMD / 32, DIMD / 32), 256, 0, stream>>>(
      w_qkv, wqkvT, DIMD, 3 * DIMD);
  transpose_f32_f16<<<dim3(DIMD / 32, DIMD / 32), 256, 0, stream>>>(
      w_proj, wprojT, DIMD, DIMD);

  // 2) QKV projection: [8192,1024] x [1024,3072] + b -> Qh (x1/32), Kh, VhT.
  gemm_wmma<0><<<dim3(3 * DIMD / 128, BSEQ / 128, 1), 256, 0, stream>>>(
      xh, wqkvT, nullptr, nullptr, b_qkv, 3 * DIMD, DIMD,
      0L, 0L, 0L, Qh, Kh, VhT);

  // 3) Scores = (Q/32) @ K^T per batch: [2048,1024] x [2048,1024]^T.
  gemm_wmma<1><<<dim3(SEQ / 128, SEQ / 128, BATCH), 256, 0, stream>>>(
      Qh, Kh, Sc, nullptr, nullptr, SEQ, DIMD,
      (long)SEQ * DIMD, (long)SEQ * DIMD, (long)SEQ * SEQ,
      nullptr, nullptr, nullptr);

  // 4) Row softmax -> P (f16).
  softmax_rows<<<dim3(BATCH * SEQ), 256, 0, stream>>>(Sc, P);

  // 5) O = P @ V per batch: [2048,2048] x [1024,2048]^T (V pre-transposed).
  gemm_wmma<2><<<dim3(DIMD / 128, SEQ / 128, BATCH), 256, 0, stream>>>(
      P, VhT, nullptr, Oh, nullptr, DIMD, SEQ,
      (long)SEQ * SEQ, (long)DIMD * SEQ, (long)SEQ * DIMD,
      nullptr, nullptr, nullptr);

  // 6) Output projection: [8192,1024] x [1024,1024]^T + b -> d_out (fp32).
  gemm_wmma<3><<<dim3(DIMD / 128, BSEQ / 128, 1), 256, 0, stream>>>(
      Oh, wprojT, out, nullptr, b_proj, DIMD, DIMD,
      0L, 0L, 0L, nullptr, nullptr, nullptr);
}
